// MultiTimeAttention_10866267259560
// MI455X (gfx1250) — matrix operand order repaired and checked by reading, not compiled, not run on hardware
//
#include <hip/hip_runtime.h>

// ---------------------------------------------------------------------------
// Multi-time attention, fused flash-attention style on CDNA5 (gfx1250):
//  - all GEMMs on v_wmma_f32_16x16x32_f16 (wave32)
//  - K/V tiles staged via GLOBAL_LOAD_ASYNC_TO_LDS_B128 (ASYNCcnt), double
//    buffered against WMMA compute
// ---------------------------------------------------------------------------

typedef _Float16 v16h __attribute__((ext_vector_type(16)));
typedef _Float16 v8h  __attribute__((ext_vector_type(8)));
typedef float    v8f  __attribute__((ext_vector_type(8)));
typedef int      v4i  __attribute__((vector_size(16)));  // matches builtin param

#define WMMA_F16(a, b, c) \
  __builtin_amdgcn_wmma_f32_16x16x32_f16(false, (a), false, (b), (short)0, (c), false, false)

#if __has_builtin(__builtin_amdgcn_global_load_async_to_lds_b128)
#define HAVE_ASYNC 1
#else
#define HAVE_ASYNC 0
#endif

#if HAVE_ASYNC
#if __has_builtin(__builtin_amdgcn_s_wait_asynccnt)
#define WAIT_ASYNC(n) __builtin_amdgcn_s_wait_asynccnt(n)
#else
#define WAIT_ASYNC(n) asm volatile("s_wait_asynccnt %0" ::"i"(n) : "memory")
#endif
#else
#define WAIT_ASYNC(n) do {} while (0)
#endif

// Copy 16 bytes global -> LDS. Async (no VGPR round trip) when available.
__device__ __forceinline__ void copy_b128(const _Float16* gsrc, _Float16* ldst) {
#if HAVE_ASYNC
  // Builtin wants (AS1 v4i*, AS3 v4i*, imm offset, imm cpol).
  // AS1 pointers are numerically identical to flat pointers; flat LDS
  // addresses carry the LDS byte offset in their low 32 bits.
  __builtin_amdgcn_global_load_async_to_lds_b128(
      (__attribute__((address_space(1))) v4i*)(uintptr_t)gsrc,
      (__attribute__((address_space(3))) v4i*)(uint32_t)(uintptr_t)ldst,
      0, 0);
#else
  *(v8h*)ldst = *(const v8h*)gsrc;
#endif
}

// Problem constants (match reference)
constexpr int B_  = 8;
constexpr int LQ  = 2048;
constexpr int LK  = 2048;
constexpr int ET  = 128;   // EMBED_TIME
constexpr int H   = 8;     // NUM_HEADS
constexpr int EK  = 16;    // head dim for q/k
constexpr int D   = 128;   // value dim (INPUT_DIM)
constexpr int NH  = 128;   // NHIDDEN
constexpr int QBLK = 64;   // q rows per workgroup (4 waves x 16)
constexpr int KBLK = 128;  // k rows per inner tile

// Async ops issued per wave per tile: K = 128*16/8/128 = 2, V = 128*128/8/128 = 16
constexpr int ASYNC_PER_TILE = 18;

// Workspace layout (units: _Float16 elements). Total ~46.5 MB.
constexpr size_t OFF_Q16 = 0;                                    // (B,H,LQ,EK)
constexpr size_t OFF_K16 = OFF_Q16 + (size_t)B_ * H * LQ * EK;   // (B,H,LK,EK)
constexpr size_t OFF_VT  = OFF_K16 + (size_t)B_ * H * LK * EK;   // (B,D,LK)
constexpr size_t OFF_X16 = OFF_VT + (size_t)B_ * D * LK;         // (B*LQ, H*D)
constexpr size_t OFF_WQT = OFF_X16 + (size_t)B_ * LQ * H * D;    // (128n,128k)
constexpr size_t OFF_WKT = OFF_WQT + (size_t)ET * ET;            // (128n,128k)
constexpr size_t OFF_WOT = OFF_WKT + (size_t)ET * ET;            // (128n,1024k)

// ---------------------------------------------------------------------------
// Prep kernels
// ---------------------------------------------------------------------------
__global__ void k_transpose_f32_to_f16(const float* __restrict__ src,
                                       _Float16* __restrict__ dst,
                                       int K, int N) {
  int idx = blockIdx.x * blockDim.x + threadIdx.x;
  if (idx < K * N) {
    int k = idx / N, n = idx % N;
    dst[(size_t)n * K + k] = (_Float16)src[idx];
  }
}

// value (B, LK, D) f32 -> vT (B, D, LK) f16 so V tiles are row-contiguous
__global__ void k_vtranspose(const float* __restrict__ src,
                             _Float16* __restrict__ dst) {
  int idx = blockIdx.x * blockDim.x + threadIdx.x;
  if (idx < B_ * LK * D) {
    int b = idx / (LK * D), r = idx % (LK * D);
    int k = r / D, d = r % D;
    dst[((size_t)b * D + d) * LK + k] = (_Float16)src[idx];
  }
}

// ---------------------------------------------------------------------------
// Input projection: out16(B,H,L,EK) = reshape( (X @ W + b) * scale )
// ---------------------------------------------------------------------------
__global__ __launch_bounds__(256) void k_project(
    const float* __restrict__ X,      // (B*L, 128) fp32
    const _Float16* __restrict__ WT,  // (128 n, 128 k) f16, transposed
    const float* __restrict__ bias,   // (128)
    _Float16* __restrict__ out,       // (B,H,L,EK)
    float scale) {
  const int lane = threadIdx.x & 31;
  const int wave = threadIdx.x >> 5;
  const int hi = (lane >> 4) & 1;
  const int ln = lane & 15;
  const int rowBase = blockIdx.x * 16;
  const int n0 = wave * 16;

  v8f acc = {};
  const float* arow = X + (size_t)(rowBase + ln) * ET;
  const _Float16* brow = WT + (size_t)(n0 + ln) * ET;
#pragma unroll
  for (int k0 = 0; k0 < ET; k0 += 32) {
    v16h a, bf;
    const float* ap = arow + k0 + hi * 8;
#pragma unroll
    for (int i = 0; i < 8; ++i) {
      a[i]     = (_Float16)ap[i];        // k = k0 + hi*8 + i
      a[8 + i] = (_Float16)ap[16 + i];   // k = k0 + 16 + hi*8 + i
    }
    v8h b0 = *(const v8h*)(brow + k0 + hi * 8);
    v8h b1 = *(const v8h*)(brow + k0 + 16 + hi * 8);
#pragma unroll
    for (int i = 0; i < 8; ++i) { bf[i] = b0[i]; bf[8 + i] = b1[i]; }
    acc = WMMA_F16(a, bf, acc);
  }

  const int n = n0 + ln;
  const int h = n >> 4, e = n & 15;
  const float bn = bias[n];
#pragma unroll
  for (int j = 0; j < 8; ++j) {
    int r = rowBase + j + hi * 8;       // row in (B*L)
    int b = r / LQ, l = r % LQ;
    float v = (acc[j] + bn) * scale;
    out[(((size_t)b * H + h) * LQ + l) * EK + e] = (_Float16)v;
  }
}

// ---------------------------------------------------------------------------
// Fused attention: per (b,h), 64-row Q block, online softmax over LK.
// 4 waves/block; K/V tiles double-buffered via async LDS loads.
// ---------------------------------------------------------------------------
__global__ __launch_bounds__(128) void k_attn(
    const _Float16* __restrict__ q16,  // (B,H,LQ,EK), pre-scaled by 1/sqrt(EK)
    const _Float16* __restrict__ k16,  // (B,H,LK,EK)
    const _Float16* __restrict__ vT,   // (B,D,LK)
    _Float16* __restrict__ x16) {      // (B*LQ, H*D)
  __shared__ _Float16 lds_k[2][KBLK * EK];     // [buf][key][e]    4 KB x2
  __shared__ _Float16 lds_v[2][D * KBLK];      // [buf][d][key]   32 KB x2
  __shared__ _Float16 lds_p[4 * 16 * KBLK];    // per-wave P      16 KB

  const int bh = blockIdx.x;
  const int b = bh / H, h = bh % H;
  const int q0 = blockIdx.y * QBLK;
  const int tid = threadIdx.x;
  const int lane = tid & 31, wave = tid >> 5;
  const int hi = (lane >> 4) & 1, ln = lane & 15;

  const _Float16* kbase = k16 + ((size_t)b * H + h) * LK * EK;
  const _Float16* vbase = vT + (size_t)b * D * LK;

  // Q fragment, contraction padded 16 -> 32 (elements 8..15 are k>=16 -> 0)
  v16h qa = {};
  {
    const _Float16* qp =
        q16 + ((((size_t)b * H + h) * LQ) + q0 + wave * 16 + ln) * EK + hi * 8;
    v8h qv = *(const v8h*)qp;
#pragma unroll
    for (int i = 0; i < 8; ++i) qa[i] = qv[i];
  }

  v8f o[8];
#pragma unroll
  for (int dt = 0; dt < 8; ++dt) o[dt] = {};
  float mrow[8], lrow[8];
#pragma unroll
  for (int j = 0; j < 8; ++j) { mrow[j] = -1e30f; lrow[j] = 0.f; }

  _Float16* myp = lds_p + wave * 16 * KBLK;

  auto issue_tile = [&](int kt, int buf) {
    // K tile: KBLK x EK halves, contiguous
#pragma unroll
    for (int c = tid; c < KBLK * EK / 8; c += 128)
      copy_b128(kbase + (size_t)kt * EK + c * 8, &lds_k[buf][c * 8]);
    // V tile: D rows of KBLK halves from vT (row stride LK)
#pragma unroll
    for (int c = tid; c < D * KBLK / 8; c += 128) {
      int d = c >> 4;             // c / (KBLK/8)
      int x8 = (c & 15) << 3;
      copy_b128(vbase + (size_t)d * LK + kt + x8, &lds_v[buf][d * KBLK + x8]);
    }
  };

  issue_tile(0, 0);
  int cur = 0;
  for (int kt = 0; kt < LK; kt += KBLK) {
    const bool have_next = (kt + KBLK) < LK;
    if (have_next) {
      issue_tile(kt + KBLK, cur ^ 1);
      WAIT_ASYNC(ASYNC_PER_TILE);  // tile kt has landed (async in-order)
    } else {
      WAIT_ASYNC(0);
    }
    __syncthreads();

    const _Float16* kb_ = lds_k[cur];
    const _Float16* vb_ = lds_v[cur];

    // S = Q K^T for 8 N-tiles of 16 keys
    v8f s[8];
#pragma unroll
    for (int t = 0; t < 8; ++t) {
      v16h kb = {};
      {  // elements 8..15 carry padded k>=16 -> stay zero
        const _Float16* kp = kb_ + (t * 16 + ln) * EK + hi * 8;
        v8h a0 = *(const v8h*)kp;
#pragma unroll
        for (int i = 0; i < 8; ++i) { kb[i] = a0[i]; }
      }
      v8f z = {};
      s[t] = WMMA_F16(qa, kb, z);
    }

    // Online softmax per row (rows live across 16-lane halves)
#pragma unroll
    for (int j = 0; j < 8; ++j) {
      float cur_m = s[0][j];
#pragma unroll
      for (int t = 1; t < 8; ++t) cur_m = fmaxf(cur_m, s[t][j]);
#pragma unroll
      for (int msk = 1; msk <= 8; msk <<= 1)
        cur_m = fmaxf(cur_m, __shfl_xor(cur_m, msk, 32));
      float mnew = fmaxf(mrow[j], cur_m);
      float sc = __expf(mrow[j] - mnew);
      mrow[j] = mnew;
      lrow[j] *= sc;
#pragma unroll
      for (int dt = 0; dt < 8; ++dt) o[dt][j] *= sc;
      float rs = 0.f;
#pragma unroll
      for (int t = 0; t < 8; ++t) {
        float p = __expf(s[t][j] - mnew);
        s[t][j] = p;
        rs += p;
      }
#pragma unroll
      for (int msk = 1; msk <= 8; msk <<= 1) rs += __shfl_xor(rs, msk, 32);
      lrow[j] += rs;
    }

    // P: C-layout regs -> A-layout via per-wave LDS bounce
#pragma unroll
    for (int t = 0; t < 8; ++t)
#pragma unroll
      for (int j = 0; j < 8; ++j)
        myp[(j + hi * 8) * KBLK + t * 16 + ln] = (_Float16)s[t][j];

    // O += P @ V   (contraction over 128 keys: 4 K-steps of 32)
#pragma unroll
    for (int kk = 0; kk < 4; ++kk) {
      v16h pa;
      const _Float16* pp = myp + ln * KBLK + kk * 32 + hi * 8;
      v8h p0 = *(const v8h*)pp;
      v8h p1 = *(const v8h*)(pp + 16);
#pragma unroll
      for (int i = 0; i < 8; ++i) { pa[i] = p0[i]; pa[8 + i] = p1[i]; }
#pragma unroll
      for (int dt = 0; dt < 8; ++dt) {
        const _Float16* vp = vb_ + (dt * 16 + ln) * KBLK + kk * 32 + hi * 8;
        v8h b0 = *(const v8h*)vp;
        v8h b1 = *(const v8h*)(vp + 16);
        v16h vb;
#pragma unroll
        for (int i = 0; i < 8; ++i) { vb[i] = b0[i]; vb[8 + i] = b1[i]; }
        o[dt] = WMMA_F16(pa, vb, o[dt]);
      }
    }

    __syncthreads();  // all waves done with buffer `cur` before it is refilled
    cur ^= 1;
  }

  // Normalize and emit x in (B*LQ, H*D) f16 for the output projection
#pragma unroll
  for (int j = 0; j < 8; ++j) lrow[j] = 1.0f / lrow[j];
  const int qrow = q0 + wave * 16;
#pragma unroll
  for (int dt = 0; dt < 8; ++dt)
#pragma unroll
    for (int j = 0; j < 8; ++j) {
      int r = qrow + j + hi * 8;
      x16[(((size_t)b * LQ + r) * (H * D)) + h * D + dt * 16 + ln] =
          (_Float16)(o[dt][j] * lrow[j]);
    }
}

// ---------------------------------------------------------------------------
// Output projection: out(B*LQ,128) = x16(B*LQ,1024) @ Wo(1024,128) + bo
// ---------------------------------------------------------------------------
__global__ __launch_bounds__(256) void k_outproj(
    const _Float16* __restrict__ x16,  // (B*LQ, 1024)
    const _Float16* __restrict__ WoT,  // (128 n, 1024 k)
    const float* __restrict__ bo,      // (128)
    float* __restrict__ out) {         // (B*LQ, 128) fp32
  const int lane = threadIdx.x & 31;
  const int wave = threadIdx.x >> 5;
  const int hi = (lane >> 4) & 1, ln = lane & 15;
  const int rowBase = blockIdx.x * 16;
  const int n0 = wave * 16;
  constexpr int KD = H * D;  // 1024

  v8f acc = {};
  const _Float16* arow = x16 + (size_t)(rowBase + ln) * KD;
  const _Float16* brow = WoT + (size_t)(n0 + ln) * KD;
  for (int k0 = 0; k0 < KD; k0 += 32) {
    v16h a, bf;
    v8h a0 = *(const v8h*)(arow + k0 + hi * 8);
    v8h a1 = *(const v8h*)(arow + k0 + 16 + hi * 8);
    v8h b0 = *(const v8h*)(brow + k0 + hi * 8);
    v8h b1 = *(const v8h*)(brow + k0 + 16 + hi * 8);
#pragma unroll
    for (int i = 0; i < 8; ++i) {
      a[i] = a0[i]; a[8 + i] = a1[i];
      bf[i] = b0[i]; bf[8 + i] = b1[i];
    }
    acc = WMMA_F16(a, bf, acc);
  }
  const float bn = bo[n0 + ln];
#pragma unroll
  for (int j = 0; j < 8; ++j)
    out[(size_t)(rowBase + j + hi * 8) * NH + n0 + ln] = acc[j] + bn;
}

// ---------------------------------------------------------------------------
extern "C" void kernel_launch(void* const* d_in, const int* in_sizes, int n_in,
                              void* d_out, int out_size, void* d_ws,
                              size_t ws_size, hipStream_t stream) {
  const float* query = (const float*)d_in[0];
  const float* key   = (const float*)d_in[1];
  const float* value = (const float*)d_in[2];
  const float* Wq    = (const float*)d_in[3];
  const float* bq    = (const float*)d_in[4];
  const float* Wk    = (const float*)d_in[5];
  const float* bk    = (const float*)d_in[6];
  const float* Wo    = (const float*)d_in[7];
  const float* bo    = (const float*)d_in[8];
  float* out = (float*)d_out;

  _Float16* ws  = (_Float16*)d_ws;
  _Float16* q16 = ws + OFF_Q16;
  _Float16* k16 = ws + OFF_K16;
  _Float16* vT  = ws + OFF_VT;
  _Float16* x16 = ws + OFF_X16;
  _Float16* wqT = ws + OFF_WQT;
  _Float16* wkT = ws + OFF_WKT;
  _Float16* woT = ws + OFF_WOT;

  // Weight transpose/convert + value transpose
  k_transpose_f32_to_f16<<<(ET * ET + 255) / 256, 256, 0, stream>>>(Wq, wqT, ET, ET);
  k_transpose_f32_to_f16<<<(ET * ET + 255) / 256, 256, 0, stream>>>(Wk, wkT, ET, ET);
  k_transpose_f32_to_f16<<<(H * D * NH + 255) / 256, 256, 0, stream>>>(Wo, woT, H * D, NH);
  k_vtranspose<<<(B_ * LK * D + 255) / 256, 256, 0, stream>>>(value, vT);

  // q/k projections (q pre-scaled by 1/sqrt(EK) = 0.25)
  k_project<<<(B_ * LQ) / 16, 256, 0, stream>>>(query, wqT, bq, q16, 0.25f);
  k_project<<<(B_ * LK) / 16, 256, 0, stream>>>(key, wkT, bk, k16, 1.0f);

  // fused attention
  dim3 ag(B_ * H, LQ / QBLK);
  k_attn<<<ag, 128, 0, stream>>>(q16, k16, vT, x16);

  // output projection
  k_outproj<<<(B_ * LQ) / 16, 256, 0, stream>>>(x16, woT, bo, out);
}
